// GCNet_15539191677813
// MI455X (gfx1250) — matrix operand order repaired and checked by reading, not compiled, not run on hardware
//
#include <hip/hip_runtime.h>
#include <hip/hip_bf16.h>

typedef __attribute__((ext_vector_type(16))) __bf16 v16bf;
typedef __attribute__((ext_vector_type(8)))  float  v8f;

#define N_NODES 256
#define N_EDGES 2048
#define C_NODE  128
#define CE_IN   160
#define HW      49

__device__ __forceinline__ unsigned short f2bf(float f) {
  unsigned u = __float_as_uint(f);
  unsigned r = (u >> 16) & 1u;
  u += 0x7fffu + r;               // round-to-nearest-even
  return (unsigned short)(u >> 16);
}
__device__ __forceinline__ float bf2f(unsigned short h) {
  return __uint_as_float(((unsigned)h) << 16);
}

// ---------------- f32 NCHW -> bf16 [b][49][C] ----------------
__global__ void cvt_act(const float* __restrict__ in,
                        unsigned short* __restrict__ out, int B, int Ch) {
  int total = B * Ch * HW;
  for (int i = blockIdx.x * blockDim.x + threadIdx.x; i < total;
       i += gridDim.x * blockDim.x) {
    int p = i % HW;
    int t = i / HW;
    int c = t % Ch;
    int b = t / Ch;
    out[((size_t)b * HW + p) * Ch + c] = f2bf(in[i]);
  }
}

// ---------------- pack OIHW f32 weights -> bf16 [tap][cb][O][32] ----------------
__global__ void pack_w(const float* __restrict__ w,
                       unsigned short* __restrict__ out, int O, int I, int CB) {
  int total = 9 * CB * O * 32;
  for (int i = blockIdx.x * blockDim.x + threadIdx.x; i < total;
       i += gridDim.x * blockDim.x) {
    int k = i & 31;
    int t = i >> 5;
    int o = t % O; t /= O;
    int cb = t % CB;
    int tap = t / CB;
    int ci = cb * 32 + k;
    float v = (ci < I) ? w[((size_t)o * I + ci) * 9 + tap] : 0.f;
    out[i] = f2bf(v);
  }
}

// ---------------- LDS staging of one channel range of one segment ----------------
__device__ __forceinline__ void stage_seg(unsigned short* lds, int ldsC, int b,
                                          const unsigned short* __restrict__ src,
                                          const int* __restrict__ idx, int cs,
                                          int off, int c_lo, int c_hi, int tid) {
  if (!src) return;
  int lo = off > c_lo ? off : c_lo;
  int hi = (off + cs) < c_hi ? (off + cs) : c_hi;
  if (lo >= hi) return;
  int bidx = idx ? idx[b] : b;
  int nc = hi - lo;                 // even
  int ndw = nc >> 1;
  const unsigned short* sp = src + (size_t)bidx * HW * cs + (lo - off);
  unsigned int* ldw = (unsigned int*)lds;
  int ldsCdw = ldsC >> 1;
  int cbase = (lo - c_lo) >> 1;
  for (int i = tid; i < HW * ndw; i += 256) {
    int p = i / ndw, d = i - p * ndw;
    int gy = p / 7 + 1, gx = p % 7 + 1;
    unsigned int val = *(const unsigned int*)(sp + (size_t)p * cs + 2 * d);
    ldw[(gy * 9 + gx) * ldsCdw + cbase + d] = val;
  }
}

// ---------------- implicit-GEMM 3x3 conv (pad 1) via v_wmma_f32_16x16x32_bf16 ----
// One block per batch element; 8 wave32 per block; wave w owns couts [16w,16w+16).
__global__ __launch_bounds__(256) void conv3x3_wmma(
    const unsigned short* __restrict__ s0, const int* __restrict__ i0, int c0,
    const unsigned short* __restrict__ s1, const int* __restrict__ i1, int c1,
    const unsigned short* __restrict__ s2, const int* __restrict__ i2, int c2,
    int cin_pad, const unsigned short* __restrict__ wpack,
    const float* __restrict__ bias, int cout, int relu,
    const unsigned short* __restrict__ resid,   // [b][49][cout] bf16 or null
    unsigned short* __restrict__ out_b,         // [b][49][cout] bf16 or null
    float* __restrict__ out_f)                  // [b][cout][49] f32 or null
{
  extern __shared__ char smem_raw[];
  unsigned short* lds = (unsigned short*)smem_raw;
  const int tid = threadIdx.x;
  const int b = blockIdx.x;
  const int ldsC = cin_pad < 256 ? cin_pad : 256;
  const int nCBtot = cin_pad >> 5;

  const int lane = tid & 31;
  const int wid = tid >> 5;
  const int nl = lane & 15;
  const int hi = lane >> 4;
  const bool active = (wid * 16) < cout;
  const int co = wid * 16 + nl;     // A-matrix row (M)
  const int cob = wid * 16 + hi * 8;// C/D cout base for this lane's 8 VGPRs

  v8f acc[4];
  if (active) {
    #pragma unroll
    for (int pt = 0; pt < 4; ++pt)
      #pragma unroll
      for (int r = 0; r < 8; ++r) acc[pt][r] = bias[cob + r];
  }

  int pos4[4], rowbase[4];
  bool pv[4];
  #pragma unroll
  for (int pt = 0; pt < 4; ++pt) {
    int p = pt * 16 + nl;
    pos4[pt] = p;
    pv[pt] = (p < HW);
    int y = p / 7, xx = p - y * 7;
    rowbase[pt] = y * 9 + xx;       // row for tap(0,0); + ky*9 + kx per tap
  }

  for (int coff = 0; coff < cin_pad; coff += ldsC) {
    int cc = cin_pad - coff;
    if (cc > ldsC) cc = ldsC;
    __syncthreads();                // previous chunk's compute done
    {                               // zero 82 rows (incl. dummy row 81)
      unsigned int* ldw = (unsigned int*)lds;
      int tot = (82 * ldsC) >> 1;
      for (int i = tid; i < tot; i += 256) ldw[i] = 0u;
    }
    __syncthreads();
    stage_seg(lds, ldsC, b, s0, i0, c0, 0,        coff, coff + cc, tid);
    stage_seg(lds, ldsC, b, s1, i1, c1, c0,       coff, coff + cc, tid);
    stage_seg(lds, ldsC, b, s2, i2, c2, c0 + c1,  coff, coff + cc, tid);
    __syncthreads();

    if (active) {
      int nCBl = cc >> 5;
      for (int tap = 0; tap < 9; ++tap) {
        int ky = tap / 3, kx = tap - ky * 3;
        int roff[4];
        #pragma unroll
        for (int pt = 0; pt < 4; ++pt)
          roff[pt] = (pv[pt] ? (rowbase[pt] + ky * 9 + kx) : 81) * ldsC;
        const unsigned short* wt =
            wpack + (((size_t)tap * nCBtot + (coff >> 5)) * cout + co) * 32;
        for (int cb = 0; cb < nCBl; ++cb) {
          union { uint4 u[2]; v16bf v; } A;
          const uint4* ap = (const uint4*)(wt + (size_t)cb * cout * 32);
          A.u[0] = ap[hi];          // K = hi*8 .. hi*8+7
          A.u[1] = ap[2 + hi];      // K = 16+hi*8 .. +7
          #pragma unroll
          for (int pt = 0; pt < 4; ++pt) {
            union { uint4 u[2]; v16bf v; } Bm;
            const uint4* bp = (const uint4*)(lds + roff[pt] + cb * 32);
            Bm.u[0] = bp[hi];       // K = hi*8 .. hi*8+7
            Bm.u[1] = bp[2 + hi];   // K = 16+hi*8 .. +7
            acc[pt] = __builtin_amdgcn_wmma_f32_16x16x32_bf16(
                false, A.v, false, Bm.v, (short)0, acc[pt], false, false);
          }
        }
      }
    }
  }

  if (active) {
    #pragma unroll
    for (int pt = 0; pt < 4; ++pt) {
      if (!pv[pt]) continue;
      int p = pos4[pt];
      float v[8];
      #pragma unroll
      for (int r = 0; r < 8; ++r) v[r] = acc[pt][r];
      if (resid) {
        uint4 rv = *(const uint4*)(resid + ((size_t)b * HW + p) * cout + cob);
        const unsigned short* rh = (const unsigned short*)&rv;
        #pragma unroll
        for (int r = 0; r < 8; ++r) v[r] += bf2f(rh[r]);
      }
      if (relu) {
        #pragma unroll
        for (int r = 0; r < 8; ++r) v[r] = v[r] > 0.f ? v[r] : 0.f;
      }
      if (out_b) {
        unsigned short h[8];
        #pragma unroll
        for (int r = 0; r < 8; ++r) h[r] = f2bf(v[r]);
        *(uint4*)(out_b + ((size_t)b * HW + p) * cout + cob) = *(const uint4*)h;
      }
      if (out_f) {
        #pragma unroll
        for (int r = 0; r < 8; ++r)
          out_f[((size_t)b * cout + cob + r) * HW + p] = v[r];
      }
    }
  }
}

// ---------------- attention: theta/phi (pooled conv), softmax, 16x16 einsum ----
__global__ __launch_bounds__(128) void attn_kernel(
    const unsigned short* __restrict__ m_b,   // [E][49][128]
    const unsigned short* __restrict__ t_b,   // [E][49][16]
    const float* __restrict__ w_t, const float* __restrict__ b_t,
    const float* __restrict__ w_p, const float* __restrict__ b_p,
    unsigned short* __restrict__ tt_b)        // [E][49][16]
{
  int b = blockIdx.x;
  int tid = threadIdx.x;
  __shared__ float S[C_NODE * 9];
  __shared__ float theta[16], phi[16];
  __shared__ float Wm[16][16];
  {
    int ci = tid;                              // 128 threads = 128 channels
    float mv[HW];
    for (int p = 0; p < HW; ++p)
      mv[p] = bf2f(m_b[((size_t)b * HW + p) * C_NODE + ci]);
    #pragma unroll
    for (int ky = 0; ky < 3; ++ky)
      #pragma unroll
      for (int kx = 0; kx < 3; ++kx) {
        float s = 0.f;
        for (int y = 0; y < 5; ++y)
          for (int x = 0; x < 5; ++x) s += mv[(y + ky) * 7 + (x + kx)];
        S[ci * 9 + ky * 3 + kx] = s;
      }
  }
  __syncthreads();
  if (tid < 32) {
    int o = tid & 15;
    const float* wv = (tid < 16 ? w_t : w_p) + (size_t)o * C_NODE * 9;
    float acc = 0.f;
    for (int i = 0; i < C_NODE * 9; ++i) acc += wv[i] * S[i];
    acc = acc * (1.f / 25.f) + (tid < 16 ? b_t[o] : b_p[o]);
    if (tid < 16) theta[o] = acc; else phi[o] = acc;
  }
  __syncthreads();
  if (tid < 16) {
    float th = theta[tid];
    float e[16];
    float mx = -3.4e38f;
    #pragma unroll
    for (int j = 0; j < 16; ++j) { float v = th * phi[j]; e[j] = v; mx = v > mx ? v : mx; }
    float s = 0.f;
    #pragma unroll
    for (int j = 0; j < 16; ++j) { float v = expf(e[j] - mx); e[j] = v; s += v; }
    float inv = 1.f / s;
    #pragma unroll
    for (int j = 0; j < 16; ++j) Wm[tid][j] = e[j] * inv;
  }
  __syncthreads();
  for (int o = tid; o < 16 * HW; o += 128) {
    int i = o / HW, p = o - i * HW;
    float a = 0.f;
    #pragma unroll
    for (int j = 0; j < 16; ++j)
      a += Wm[i][j] * bf2f(t_b[((size_t)b * HW + p) * 16 + j]);
    tt_b[((size_t)b * HW + p) * 16 + i] = f2bf(a);
  }
}

// ---------------- deterministic segment-mean over edges -> [n][49][128] bf16 ----
__global__ __launch_bounds__(256) void aggregate_mean(
    const unsigned short* __restrict__ m,     // [E][49][128] (post-attention)
    const int* __restrict__ col,
    unsigned short* __restrict__ mean_b)      // [N][49][128]
{
  int n = blockIdx.x;
  __shared__ int list[N_EDGES];
  __shared__ int nm_s;
  if (threadIdx.x == 0) {
    int c = 0;
    for (int e = 0; e < N_EDGES; ++e)
      if (col[e] == n) list[c++] = e;
    nm_s = c;
  }
  __syncthreads();
  int nm = nm_s;
  float inv = 1.f / (float)(nm > 1 ? nm : 1);
  for (int i = threadIdx.x; i < HW * C_NODE; i += 256) {
    float a = 0.f;
    for (int k = 0; k < nm; ++k)
      a += bf2f(m[(size_t)list[k] * (HW * C_NODE) + i]);
    mean_b[(size_t)n * (HW * C_NODE) + i] = f2bf(a * inv);
  }
}

extern "C" void kernel_launch(void* const* d_in, const int* in_sizes, int n_in,
                              void* d_out, int out_size, void* d_ws, size_t ws_size,
                              hipStream_t stream) {
  const float* x    = (const float*)d_in[0];
  const float* ea   = (const float*)d_in[1];
  const int*   eidx = (const int*)d_in[2];
  const float* w_e1 = (const float*)d_in[3];  const float* b_e1 = (const float*)d_in[4];
  const float* w_e2 = (const float*)d_in[5];  const float* b_e2 = (const float*)d_in[6];
  const float* w_m1 = (const float*)d_in[7];  const float* b_m1 = (const float*)d_in[8];
  const float* w_m2 = (const float*)d_in[9];  const float* b_m2 = (const float*)d_in[10];
  const float* w_t  = (const float*)d_in[11]; const float* b_t  = (const float*)d_in[12];
  const float* w_p  = (const float*)d_in[13]; const float* b_p  = (const float*)d_in[14];
  const float* w_f  = (const float*)d_in[15]; const float* b_f  = (const float*)d_in[16];
  const float* w_g  = (const float*)d_in[17]; const float* b_g  = (const float*)d_in[18];
  const float* w_n1 = (const float*)d_in[19]; const float* b_n1 = (const float*)d_in[20];
  const float* w_n2 = (const float*)d_in[21]; const float* b_n2 = (const float*)d_in[22];
  const int* row = eidx;
  const int* col = eidx + N_EDGES;

  char* wsb = (char*)d_ws;
  size_t off = 0;
  auto alloc = [&](size_t bytes) -> void* {
    void* p = wsb + off;
    off = (off + bytes + 255) & ~(size_t)255;
    return p;
  };
  unsigned short* xall  = (unsigned short*)alloc((size_t)N_NODES * HW * C_NODE * 2);
  unsigned short* eab   = (unsigned short*)alloc((size_t)N_EDGES * HW * CE_IN * 2);
  unsigned short* e1b   = (unsigned short*)alloc((size_t)N_EDGES * HW * C_NODE * 2);
  unsigned short* ebf   = (unsigned short*)alloc((size_t)N_EDGES * HW * C_NODE * 2);
  unsigned short* m1b   = (unsigned short*)alloc((size_t)N_EDGES * HW * C_NODE * 2);
  unsigned short* mb    = (unsigned short*)alloc((size_t)N_EDGES * HW * C_NODE * 2);
  unsigned short* tb    = (unsigned short*)alloc((size_t)N_EDGES * HW * 16 * 2);
  unsigned short* ttb   = (unsigned short*)alloc((size_t)N_EDGES * HW * 16 * 2);
  unsigned short* mfb   = (unsigned short*)alloc((size_t)N_EDGES * HW * C_NODE * 2);
  unsigned short* meanb = (unsigned short*)alloc((size_t)N_NODES * HW * C_NODE * 2);
  unsigned short* n1b   = (unsigned short*)alloc((size_t)N_NODES * HW * C_NODE * 2);
  unsigned short* pk_e1 = (unsigned short*)alloc((size_t)9 * 13 * 128 * 32 * 2);
  unsigned short* pk_e2 = (unsigned short*)alloc((size_t)9 * 4  * 128 * 32 * 2);
  unsigned short* pk_m1 = (unsigned short*)alloc((size_t)9 * 8  * 128 * 32 * 2);
  unsigned short* pk_m2 = (unsigned short*)alloc((size_t)9 * 4  * 128 * 32 * 2);
  unsigned short* pk_f  = (unsigned short*)alloc((size_t)9 * 4  * 16  * 32 * 2);
  unsigned short* pk_g  = (unsigned short*)alloc((size_t)9 * 1  * 128 * 32 * 2);
  unsigned short* pk_n1 = (unsigned short*)alloc((size_t)9 * 8  * 128 * 32 * 2);
  unsigned short* pk_n2 = (unsigned short*)alloc((size_t)9 * 4  * 128 * 32 * 2);

  float* u_out = (float*)d_out;
  float* e_out = (float*)d_out + (size_t)N_NODES * C_NODE * HW;

  // activation conversion + weight packing
  cvt_act<<<512, 256, 0, stream>>>(x, xall, N_NODES, C_NODE);
  cvt_act<<<4096, 256, 0, stream>>>(ea, eab, N_EDGES, CE_IN);
  pack_w<<<512, 256, 0, stream>>>(w_e1, pk_e1, 128, 416, 13);
  pack_w<<<256, 256, 0, stream>>>(w_e2, pk_e2, 128, 128, 4);
  pack_w<<<256, 256, 0, stream>>>(w_m1, pk_m1, 128, 256, 8);
  pack_w<<<256, 256, 0, stream>>>(w_m2, pk_m2, 128, 128, 4);
  pack_w<<<64,  256, 0, stream>>>(w_f,  pk_f,  16,  128, 4);
  pack_w<<<64,  256, 0, stream>>>(w_g,  pk_g,  128, 16,  1);
  pack_w<<<256, 256, 0, stream>>>(w_n1, pk_n1, 128, 256, 8);
  pack_w<<<256, 256, 0, stream>>>(w_n2, pk_n2, 128, 128, 4);

  auto ldsBytes = [](int cin) -> size_t {
    int l = cin < 256 ? cin : 256;
    return (size_t)82 * l * 2;
  };

  // edge update: conv(relu) -> conv; e goes to d_out (f32) and ebf (bf16)
  conv3x3_wmma<<<N_EDGES, 256, ldsBytes(416), stream>>>(
      eab, nullptr, CE_IN, xall, row, C_NODE, xall, col, C_NODE,
      416, pk_e1, b_e1, 128, 1, nullptr, e1b, nullptr);
  conv3x3_wmma<<<N_EDGES, 256, ldsBytes(128), stream>>>(
      e1b, nullptr, 128, nullptr, nullptr, 0, nullptr, nullptr, 0,
      128, pk_e2, b_e2, 128, 0, nullptr, ebf, e_out);

  // message: conv(relu) -> conv -> attention
  conv3x3_wmma<<<N_EDGES, 256, ldsBytes(256), stream>>>(
      xall, row, C_NODE, ebf, nullptr, C_NODE, nullptr, nullptr, 0,
      256, pk_m1, b_m1, 128, 1, nullptr, m1b, nullptr);
  conv3x3_wmma<<<N_EDGES, 256, ldsBytes(128), stream>>>(
      m1b, nullptr, 128, nullptr, nullptr, 0, nullptr, nullptr, 0,
      128, pk_m2, b_m2, 128, 0, nullptr, mb, nullptr);
  conv3x3_wmma<<<N_EDGES, 256, ldsBytes(128), stream>>>(
      mb, nullptr, 128, nullptr, nullptr, 0, nullptr, nullptr, 0,
      128, pk_f, b_f, 16, 0, nullptr, tb, nullptr);
  attn_kernel<<<N_EDGES, 128, 0, stream>>>(mb, tb, w_t, b_t, w_p, b_p, ttb);
  conv3x3_wmma<<<N_EDGES, 256, ldsBytes(32), stream>>>(
      ttb, nullptr, 16, nullptr, nullptr, 0, nullptr, nullptr, 0,
      32, pk_g, b_g, 128, 0, mb /*residual*/, mfb, nullptr);

  // deterministic segment mean at target nodes
  aggregate_mean<<<N_NODES, 256, 0, stream>>>(mfb, col, meanb);

  // node update: conv(relu) -> conv -> u (f32)
  conv3x3_wmma<<<N_NODES, 256, ldsBytes(256), stream>>>(
      xall, nullptr, C_NODE, meanb, nullptr, C_NODE, nullptr, nullptr, 0,
      256, pk_n1, b_n1, 128, 1, nullptr, n1b, nullptr);
  conv3x3_wmma<<<N_NODES, 256, ldsBytes(128), stream>>>(
      n1b, nullptr, 128, nullptr, nullptr, 0, nullptr, nullptr, 0,
      128, pk_n2, b_n2, 128, 0, nullptr, nullptr, u_out);
}
